// Attention_48421461295932
// MI455X (gfx1250) — matrix operand order repaired and checked by reading, not compiled, not run on hardware
//
#include <hip/hip_runtime.h>

typedef __attribute__((ext_vector_type(16))) _Float16 v16h;
typedef __attribute__((ext_vector_type(8)))  float    v8f;

#define DEVINL __device__ __forceinline__

constexpr int B_    = 8;
constexpr int C_    = 512;
constexpr int N_    = 1024;   // H*W
constexpr int E_    = 512;
constexpr int HEADS = 8;
constexpr int DH    = 64;
constexpr int OC3   = 1536;   // 3*E

union FragH { uint4 u[2]; v16h h; };
union Pack8 { uint4 u; _Float16 h[8]; };

// Load a v16h fragment as two contiguous 16-byte runs (base0: elems 0..7, base1: elems 8..15)
DEVINL v16h load_frag(const _Float16* base0, const _Float16* base1) {
  FragH f;
  f.u[0] = *reinterpret_cast<const uint4*>(base0);
  f.u[1] = *reinterpret_cast<const uint4*>(base1);
  return f.h;
}

DEVINL v8f wmma_f16(v16h a, v16h b, v8f c) {
  // (neg_a, A, neg_b, B, c_mod, C, reuse_a, reuse_b)
  return __builtin_amdgcn_wmma_f32_16x16x32_f16(false, a, false, b, (short)0, c, false, false);
}

DEVINL float rmax16(float v) {   // reduce within 16-lane half (wave32)
  v = fmaxf(v, __shfl_xor(v, 1, 32));
  v = fmaxf(v, __shfl_xor(v, 2, 32));
  v = fmaxf(v, __shfl_xor(v, 4, 32));
  v = fmaxf(v, __shfl_xor(v, 8, 32));
  return v;
}
DEVINL float rsum16(float v) {
  v += __shfl_xor(v, 1, 32);
  v += __shfl_xor(v, 2, 32);
  v += __shfl_xor(v, 4, 32);
  v += __shfl_xor(v, 8, 32);
  return v;
}

// ---------------- prep: x (f32 [b][c][n]) -> xT (f16 [b][n][c]) ----------------
__global__ __launch_bounds__(256) void prep_x_kernel(const float* __restrict__ x,
                                                     _Float16* __restrict__ xT) {
  int idx = blockIdx.x * 256 + threadIdx.x;            // exactly B*C*N threads
  int n = idx % N_;
  int c = (idx / N_) % C_;
  int b = idx / (N_ * C_);
  xT[((size_t)b * N_ + n) * C_ + c] = (_Float16)x[idx];
}

// ---------------- prep: weights f32 -> f16 (same layout) ----------------
__global__ __launch_bounds__(256) void prep_w_kernel(const float* __restrict__ w_in,
                                                     const float* __restrict__ w_out,
                                                     _Float16* __restrict__ wi,
                                                     _Float16* __restrict__ wo) {
  int idx = blockIdx.x * 256 + threadIdx.x;            // OC3*C + E*C threads exactly
  if (idx < OC3 * C_) wi[idx] = (_Float16)w_in[idx];
  else {
    int j = idx - OC3 * C_;
    wo[j] = (_Float16)w_out[j];
  }
}

// ---------------- QKV: qkv[o][n] = sum_c w_in[o][c] * x[c][n] ----------------
// Wave computes 64x64 tile. Output scattered to attention-friendly layouts:
//   o <  512 : qT[b,h][n][dk]   (K-contiguous)
//   o < 1024 : kT[b,h][n][dk]
//   else     : v [b,h][dh][n]
__global__ __launch_bounds__(128) void qkv_kernel(const _Float16* __restrict__ wi,
                                                  const _Float16* __restrict__ xT,
                                                  _Float16* __restrict__ qT,
                                                  _Float16* __restrict__ kT,
                                                  _Float16* __restrict__ vv) {
  const int lane = threadIdx.x & 31;
  const int wv   = threadIdx.x >> 5;
  const int lh   = lane >> 4, ll = lane & 15;
  const int b    = blockIdx.z;
  const int m0   = blockIdx.x * 128 + (wv >> 1) * 64;
  const int n0   = blockIdx.y * 128 + (wv & 1) * 64;
  const _Float16* xb = xT + (size_t)b * N_ * C_;

  v8f acc[4][4] = {};

  for (int k0 = 0; k0 < C_; k0 += 32) {
    const _Float16* pa[4];
    const _Float16* pb[4];
    v16h af[4], bf[4];
#pragma unroll
    for (int mi = 0; mi < 4; mi++) {
      pa[mi] = wi + (size_t)(m0 + 16 * mi + ll) * C_ + k0 + 8 * lh;
      af[mi] = load_frag(pa[mi], pa[mi] + 16);
    }
#pragma unroll
    for (int ni = 0; ni < 4; ni++) {
      pb[ni] = xb + (size_t)(n0 + 16 * ni + ll) * C_ + k0 + 8 * lh;
      bf[ni] = load_frag(pb[ni], pb[ni] + 16);
    }
    if (k0 + 32 < C_) {            // single uniform branch per K-step
#pragma unroll
      for (int mi = 0; mi < 4; mi++) __builtin_prefetch(pa[mi] + 32, 0, 3);
#pragma unroll
      for (int ni = 0; ni < 4; ni++) __builtin_prefetch(pb[ni] + 32, 0, 3);
    }
#pragma unroll
    for (int mi = 0; mi < 4; mi++)
#pragma unroll
      for (int ni = 0; ni < 4; ni++)
        acc[mi][ni] = wmma_f16(af[mi], bf[ni], acc[mi][ni]);
  }

#pragma unroll
  for (int mi = 0; mi < 4; mi++) {
    const int ob = m0 + 16 * mi;            // fragment row base (never crosses a 64-boundary)
#pragma unroll
    for (int ni = 0; ni < 4; ni++) {
      const int n = n0 + 16 * ni + ll;
      if (ob < E_) {
        const int h = ob / DH, dkb = (ob % DH) + 8 * lh;
        Pack8 pk;
#pragma unroll
        for (int r = 0; r < 8; r++) pk.h[r] = (_Float16)acc[mi][ni][r];
        *reinterpret_cast<uint4*>(qT + (((size_t)(b * HEADS + h) * N_ + n) * DH + dkb)) = pk.u;
      } else if (ob < 2 * E_) {
        const int o2 = ob - E_;
        const int h = o2 / DH, dkb = (o2 % DH) + 8 * lh;
        Pack8 pk;
#pragma unroll
        for (int r = 0; r < 8; r++) pk.h[r] = (_Float16)acc[mi][ni][r];
        *reinterpret_cast<uint4*>(kT + (((size_t)(b * HEADS + h) * N_ + n) * DH + dkb)) = pk.u;
      } else {
        const int o2 = ob - 2 * E_;
        const int h = o2 / DH, dhb = (o2 % DH) + 8 * lh;
        _Float16* vp = vv + ((size_t)(b * HEADS + h) * DH) * N_;
#pragma unroll
        for (int r = 0; r < 8; r++)
          vp[(size_t)(dhb + r) * N_ + n] = (_Float16)acc[mi][ni][r];
      }
    }
  }
}

// ---------------- fused flash attention per (b, h, 16-row i-tile) ----------------
__global__ __launch_bounds__(128) void attn_kernel(const _Float16* __restrict__ qT,
                                                   const _Float16* __restrict__ kT,
                                                   const _Float16* __restrict__ vv,
                                                   _Float16* __restrict__ oT) {
  __shared__ __align__(16) _Float16 plds[4][16 * 32];   // P staging, 1KB per wave

  const int lane = threadIdx.x & 31;
  const int wv   = threadIdx.x >> 5;
  const int lh   = lane >> 4, ll = lane & 15;
  const int t  = blockIdx.x * 4 + wv;                   // tile id in [0, 4096)
  const int b  = t >> 9;                                // / (HEADS * 64)
  const int h  = (t >> 6) & 7;
  const int i0 = (t & 63) * 16;

  const _Float16* qb = qT + (size_t)(b * HEADS + h) * N_ * DH;
  const _Float16* kb = kT + (size_t)(b * HEADS + h) * N_ * DH;
  const _Float16* vb = vv + (size_t)(b * HEADS + h) * DH * N_;

  // A fragments of q^T (M=i, K=dk), dk 0..31 and 32..63
  const _Float16* qr = qb + (size_t)(i0 + ll) * DH;
  const v16h aq0 = load_frag(qr +  8 * lh,      qr + 16 + 8 * lh);
  const v16h aq1 = load_frag(qr + 32 + 8 * lh,  qr + 48 + 8 * lh);

  float mrow[8], lrow[8];
  v8f O[4] = {};
#pragma unroll
  for (int r = 0; r < 8; r++) { mrow[r] = -3.0e38f; lrow[r] = 0.0f; }

  const float scale = 0.35355339059327373f;   // heads ** -0.5

  for (int a0 = 0; a0 < N_; a0 += 32) {
    // ---- S = scale * q^T k for two 16-col tiles ----
    v8f s0 = {}, s1 = {};
    {
      const _Float16* kc = kb + (size_t)(a0 + ll) * DH + 8 * lh;
      s0 = wmma_f16(aq0, load_frag(kc,      kc + 16), s0);
      s0 = wmma_f16(aq1, load_frag(kc + 32, kc + 48), s0);
    }
    {
      const _Float16* kc = kb + (size_t)(a0 + 16 + ll) * DH + 8 * lh;
      s1 = wmma_f16(aq0, load_frag(kc,      kc + 16), s1);
      s1 = wmma_f16(aq1, load_frag(kc + 32, kc + 48), s1);
    }

    // ---- online softmax (rows live in-lane: vgpr r <-> row r + 8*lh) ----
    float p0[8], p1[8];
#pragma unroll
    for (int r = 0; r < 8; r++) {
      const float x0 = s0[r] * scale, x1 = s1[r] * scale;
      const float mx   = rmax16(fmaxf(x0, x1));
      const float mnew = fmaxf(mrow[r], mx);
      const float corr = __expf(mrow[r] - mnew);
      const float e0 = __expf(x0 - mnew), e1 = __expf(x1 - mnew);
      lrow[r] = lrow[r] * corr + rsum16(e0 + e1);
      mrow[r] = mnew;
      p0[r] = e0; p1[r] = e1;
#pragma unroll
      for (int f = 0; f < 4; f++) O[f][r] *= corr;
    }

    // ---- P: D-layout -> A-layout via LDS ----
    __syncthreads();
    _Float16* pw = plds[wv];
#pragma unroll
    for (int r = 0; r < 8; r++) {
      const int m = r + 8 * lh;
      pw[m * 32 + ll]      = (_Float16)p0[r];
      pw[m * 32 + 16 + ll] = (_Float16)p1[r];
    }
    __syncthreads();
    const _Float16* pr = pw + ll * 32 + 8 * lh;
    const v16h pA = load_frag(pr, pr + 16);

    // ---- O += P * v^T ----
#pragma unroll
    for (int f = 0; f < 4; f++) {
      const _Float16* vp = vb + (size_t)(16 * f + ll) * N_ + a0 + 8 * lh;
      O[f] = wmma_f16(pA, load_frag(vp, vp + 16), O[f]);
    }
  }

  // ---- normalize and store oT[b][n][e] (K-contiguous for the proj GEMM) ----
#pragma unroll
  for (int r = 0; r < 8; r++) lrow[r] = 1.0f / lrow[r];
#pragma unroll
  for (int f = 0; f < 4; f++)
#pragma unroll
    for (int r = 0; r < 8; r++) {
      const int n = i0 + r + 8 * lh;
      const int e = h * DH + 16 * f + ll;
      oT[((size_t)b * N_ + n) * E_ + e] = (_Float16)(O[f][r] * lrow[r]);
    }
}

// ---------------- out[c][n] = w_out[c][:] . o[:][n] + b_out[c] + x[c][n] ----------------
__global__ __launch_bounds__(128) void proj_kernel(const _Float16* __restrict__ wo,
                                                   const _Float16* __restrict__ oT,
                                                   const float* __restrict__ x,
                                                   const float* __restrict__ b_out,
                                                   float* __restrict__ out) {
  const int lane = threadIdx.x & 31;
  const int wv   = threadIdx.x >> 5;
  const int lh   = lane >> 4, ll = lane & 15;
  const int b    = blockIdx.z;
  const int m0   = blockIdx.x * 128 + (wv >> 1) * 64;
  const int n0   = blockIdx.y * 128 + (wv & 1) * 64;
  const _Float16* ob = oT + (size_t)b * N_ * E_;

  v8f acc[4][4] = {};

  for (int k0 = 0; k0 < E_; k0 += 32) {
    const _Float16* pa[4];
    const _Float16* pb[4];
    v16h af[4], bf[4];
#pragma unroll
    for (int mi = 0; mi < 4; mi++) {
      pa[mi] = wo + (size_t)(m0 + 16 * mi + ll) * E_ + k0 + 8 * lh;
      af[mi] = load_frag(pa[mi], pa[mi] + 16);
    }
#pragma unroll
    for (int ni = 0; ni < 4; ni++) {
      pb[ni] = ob + (size_t)(n0 + 16 * ni + ll) * E_ + k0 + 8 * lh;
      bf[ni] = load_frag(pb[ni], pb[ni] + 16);
    }
    if (k0 + 32 < E_) {            // single uniform branch per K-step
#pragma unroll
      for (int mi = 0; mi < 4; mi++) __builtin_prefetch(pa[mi] + 32, 0, 3);
#pragma unroll
      for (int ni = 0; ni < 4; ni++) __builtin_prefetch(pb[ni] + 32, 0, 3);
    }
#pragma unroll
    for (int mi = 0; mi < 4; mi++)
#pragma unroll
      for (int ni = 0; ni < 4; ni++)
        acc[mi][ni] = wmma_f16(af[mi], bf[ni], acc[mi][ni]);
  }

#pragma unroll
  for (int mi = 0; mi < 4; mi++)
#pragma unroll
    for (int ni = 0; ni < 4; ni++) {
      const int n = n0 + 16 * ni + ll;
#pragma unroll
      for (int r = 0; r < 8; r++) {
        const int c = m0 + 16 * mi + r + 8 * lh;
        const size_t idx = ((size_t)b * C_ + c) * N_ + n;
        out[idx] = acc[mi][ni][r] + b_out[c] + x[idx];
      }
    }
}

extern "C" void kernel_launch(void* const* d_in, const int* in_sizes, int n_in,
                              void* d_out, int out_size, void* d_ws, size_t ws_size,
                              hipStream_t stream) {
  const float* x     = (const float*)d_in[0];
  const float* w_in  = (const float*)d_in[1];
  const float* w_out = (const float*)d_in[2];
  const float* b_out = (const float*)d_in[3];
  // d_in[4] = heads (constant 8, baked in)

  char* ws = (char*)d_ws;
  _Float16* xT = (_Float16*)(ws + 0);           //  8 MB: [b][n][c]
  _Float16* wi = (_Float16*)(ws + 8388608);     //  1.5 MB
  _Float16* wo = (_Float16*)(ws + 9961472);     //  0.5 MB
  _Float16* qT = (_Float16*)(ws + 10485760);    //  8 MB: [b,h][n][dk]
  _Float16* kT = (_Float16*)(ws + 18874368);    //  8 MB: [b,h][n][dk]
  _Float16* vv = (_Float16*)(ws + 27262976);    //  8 MB: [b,h][dh][n]
  _Float16* oT = (_Float16*)(ws + 35651584);    //  8 MB: [b][n][e]
  float* out = (float*)d_out;

  prep_x_kernel<<<16384, 256, 0, stream>>>(x, xT);                 // B*C*N threads
  prep_w_kernel<<<4096, 256, 0, stream>>>(w_in, w_out, wi, wo);    // (3E+E)*C threads
  qkv_kernel<<<dim3(12, 8, 8), 128, 0, stream>>>(wi, xT, qT, kT, vv);
  attn_kernel<<<1024, 128, 0, stream>>>(qT, kT, vv, oT);
  proj_kernel<<<dim3(4, 8, 8), 128, 0, stream>>>(wo, oT, x, b_out, out);
}
